// TrunkM_82935818486342
// MI455X (gfx1250) — matrix operand order repaired and checked by reading, not compiled, not run on hardware
//
#include <hip/hip_runtime.h>

typedef __attribute__((ext_vector_type(16))) _Float16 v16h;
typedef __attribute__((ext_vector_type(8)))  _Float16 v8h;
typedef __attribute__((ext_vector_type(8)))  float    v8f;

#define C1      150
#define OHW     196      // 14x14
#define CHW     29400    // 150*196
#define DMODEL  32
#define SEQLEN  196
#define KGLOB   2352u
#define QSCALE  0.35355339059327373f   // 1/sqrt(8)

// ===================== CDNA5 TDM (tensor_load_to_lds) helpers ==============
#if defined(__has_builtin)
#if __has_builtin(__builtin_amdgcn_tensor_load_to_lds)
#define USE_TDM 1
#endif
#if __has_builtin(__builtin_amdgcn_s_wait_tensorcnt)
#define HAVE_TENSORCNT 1
#endif
#endif

__device__ __forceinline__ unsigned lds_offset(const void* p) {
  return (unsigned)(unsigned long long)(__attribute__((address_space(3))) const char*)p;
}

#ifdef USE_TDM
typedef __attribute__((ext_vector_type(4))) unsigned tdm_v4u;
typedef __attribute__((ext_vector_type(8))) int      tdm_v8i;
typedef __attribute__((ext_vector_type(4))) int      tdm_v4i;

// 1-D DMA: copy n dwords from global gsrc -> LDS byte offset lds_off.
// D# per CDNA5 ISA ch.8: group0 {count=1, lds_addr, global_addr, type=2},
// group1 {data_size=4B, tensor_dim0=n, tensor_dim1=1, tile_dim0=n, stride=n}.
__device__ __forceinline__ void tdm_copy_dwords(const void* gsrc,
                                                unsigned lds_off, unsigned n) {
  unsigned long long ga = (unsigned long long)(size_t)gsrc;
  tdm_v4u g0;
  g0[0] = 1u;                                             // count=1
  g0[1] = lds_off;                                        // lds_addr (bytes)
  g0[2] = (unsigned)ga;                                   // global_addr[31:0]
  g0[3] = (unsigned)((ga >> 32) & 0x1FFFFFFu) | (2u << 30); // addr[56:32]|type=2
  tdm_v8i g1 = {};
  g1[0] = (int)(2u << 16);                                // data_size=4 bytes
  g1[1] = (int)((n & 0xFFFFu) << 16);                     // tensor_dim0 lo16
  g1[2] = (int)(((n >> 16) & 0xFFFFu) | (1u << 16));      // td0 hi16 | td1=1
  g1[3] = (int)((n & 0xFFFFu) << 16);                     // tile_dim0
  g1[4] = 0;                                              // tile_dim1/2 = 0
  g1[5] = (int)n;                                         // dim0_stride lo32
  g1[6] = 0;
  g1[7] = 0;
  tdm_v4i z4 = {};
#if __clang_major__ >= 23
  tdm_v8i z8 = {};
  __builtin_amdgcn_tensor_load_to_lds(g0, g1, z4, z4, z8, 0);
#else
  __builtin_amdgcn_tensor_load_to_lds(g0, g1, z4, z4, 0);
#endif
}
#endif

__device__ __forceinline__ void wait_tensor0() {
#ifdef HAVE_TENSORCNT
  __builtin_amdgcn_s_wait_tensorcnt(0);
#else
  asm volatile("s_wait_tensorcnt 0" ::: "memory");
#endif
}

// ===================== WMMA fragment helpers (contiguous k = 16h + s order) ==========
__device__ __forceinline__ v8f wmma16(v16h a, v16h b, v8f c) {
  return __builtin_amdgcn_wmma_f32_16x16x32_f16(false, a, false, b, (short)0, c,
                                                false, false);
}

// A-fragment 16x32 f16 from LDS row-major (stride in halves, 16B-aligned rows).
__device__ __forceinline__ v16h frag_a(const _Float16* base, int stride, int lane) {
  int m   = lane & 15;
  int khi = (lane >> 4) << 4;                 // +16 halves for lanes 16..31
  const _Float16* p = base + m * stride + khi;
  v8h c0 = *(const v8h*)(p);
  v8h c1 = *(const v8h*)(p + 8);
  v16h a;
#pragma unroll
  for (int e = 0; e < 8; ++e) { a[e] = c0[e]; a[8 + e] = c1[e]; }
  return a;
}

// B-fragment where B[k][n] = W[row0+n][k], W row-major [rows][32] f16.
__device__ __forceinline__ v16h frag_b_rowmajor(const _Float16* W, int row0,
                                                int lane) {
  int n  = lane & 15;
  int kb = (lane >> 4) << 4;                  // +16 for lanes 16..31
  const _Float16* p = W + (row0 + n) * 32 + kb;
  v8h c0 = *(const v8h*)(p);
  v8h c1 = *(const v8h*)(p + 8);
  v16h b;
#pragma unroll
  for (int e = 0; e < 8; ++e) { b[e] = c0[e]; b[8 + e] = c1[e]; }
  return b;
}

// Scores B-fragment: B[k][n] = K[key=kt16+n][k] masked to head dims [h8,h8+8).
template <bool TAIL>
__device__ __forceinline__ v16h frag_b_score(const _Float16* Kl, int kt16,
                                             int h8, int lane) {
  int n   = lane & 15;
  int kb  = (lane >> 4) << 4;
  int key = kt16 + n;
  v16h b = {};
  bool cond = (h8 & 16) == kb;                // head slice inside this K half?
  if (TAIL) cond = cond && (key < SEQLEN);
  if (cond) {
    v8h c = *(const v8h*)(Kl + key * DMODEL + h8);
    if (h8 & 8) {
#pragma unroll
      for (int e = 0; e < 8; ++e) b[8 + e] = c[e];
    } else {
#pragma unroll
      for (int e = 0; e < 8; ++e) b[e] = c[e];
    }
  }
  return b;
}

// Ctx B-fragment from TRANSPOSED V (Vt[32 cols][224 keys], zero-padded keys).
__device__ __forceinline__ v16h frag_b_ctx(const _Float16* Vt, int c32,
                                           int h8, int lane) {
  int n   = lane & 15;
  int kb  = (lane >> 4) << 4;
  int col = h8 + (n & 7);
  const _Float16* p = Vt + col * 224 + c32 + kb;
  v8h c0 = *(const v8h*)(p);
  v8h c1 = *(const v8h*)(p + 8);
  bool ok = n < 8;
  v16h b;
#pragma unroll
  for (int e = 0; e < 8; ++e) {
    b[e]     = ok ? c0[e] : (_Float16)0.0f;
    b[8 + e] = ok ? c1[e] : (_Float16)0.0f;
  }
  return b;
}

__device__ __forceinline__ float rmax16(float v) {
  v = fmaxf(v, __shfl_xor(v, 1));
  v = fmaxf(v, __shfl_xor(v, 2));
  v = fmaxf(v, __shfl_xor(v, 4));
  v = fmaxf(v, __shfl_xor(v, 8));
  return v;
}
__device__ __forceinline__ float rsum16(float v) {
  v += __shfl_xor(v, 1);
  v += __shfl_xor(v, 2);
  v += __shfl_xor(v, 4);
  v += __shfl_xor(v, 8);
  return v;
}

// One 32-key online-softmax step. TAIL=true only for the last chunk (k0=192),
// where key>=196 masking is required.
template <bool TAIL>
__device__ __forceinline__ void flash_step(const _Float16* Kh,
                                           const _Float16* Vt, _Float16* pb,
                                           v16h aq, int k0, int h8, int lane,
                                           int n15, int hi8, float (&mr)[8],
                                           float (&lr)[8], v8f& acc) {
  v8f zc = {};
  v8f s0 = wmma16(aq, frag_b_score<TAIL>(Kh, k0,      h8, lane), zc);
  v8f s1 = wmma16(aq, frag_b_score<TAIL>(Kh, k0 + 16, h8, lane), zc);
  bool msk0 = TAIL && ((k0 + n15)      >= SEQLEN);
  bool msk1 = TAIL && ((k0 + 16 + n15) >= SEQLEN);
#pragma unroll
  for (int r = 0; r < 8; ++r) {
    float a0 = s0[r];
    float a1 = s1[r];
    if (TAIL) {
      a0 = msk0 ? -1e30f : a0;
      a1 = msk1 ? -1e30f : a1;
    }
    float rm   = rmax16(fmaxf(a0, a1));
    float mnew = fmaxf(mr[r], rm);
    float alpha = __expf(mr[r] - mnew);
    float p0 = __expf(a0 - mnew);
    float p1 = __expf(a1 - mnew);
    float rs = rsum16(p0 + p1);
    lr[r] = lr[r] * alpha + rs;
    mr[r] = mnew;
    acc[r] = acc[r] * alpha;
    int prow = r + hi8;
    pb[prow * 32 + n15]      = (_Float16)p0;
    pb[prow * 32 + 16 + n15] = (_Float16)p1;
  }
  asm volatile("s_wait_dscnt 0" ::: "memory");  // wave-local LDS ordering
  v16h ap = frag_a(pb, 32, lane);
  acc = wmma16(ap, frag_b_ctx(Vt, k0, h8, lane), acc);
}

// ===================== Kernel 1: conv 9x9 s2 p4 + ReLU + per-sample max =====
__global__ __launch_bounds__(256) void k_conv(const float* __restrict__ x,
                                              const float* __restrict__ w,
                                              float* __restrict__ A,
                                              float* __restrict__ gmax) {
  __shared__ float xin[784];
  __shared__ float wls[12150];
  __shared__ float red[256];
  const int b = blockIdx.x, tid = threadIdx.x;
  for (int i = tid; i < 784; i += 256) xin[i] = x[b * 784 + i];
#ifdef USE_TDM
  if (tid < 32) {                       // wave 0 issues one DMA descriptor
    tdm_copy_dwords(w, lds_offset(wls), 12150u);
    wait_tensor0();
  }
#else
  for (int i = tid; i < 12150; i += 256) wls[i] = w[i];
#endif
  __syncthreads();
  float lmax = 0.f;
  float* Ab = A + (size_t)b * CHW;
  for (int o = tid; o < CHW; o += 256) {
    int c  = o / OHW;
    int p  = o - c * OHW;
    int oy = p / 14;
    int ox = p - oy * 14;
    const float* wc = &wls[c * 81];
    float acc = 0.f;
#pragma unroll
    for (int ky = 0; ky < 9; ++ky) {
      int iy = oy * 2 + ky - 4;
      if (iy < 0 || iy > 27) continue;
      const float* xr = &xin[iy * 28];
      const float* wr = &wc[ky * 9];
#pragma unroll
      for (int kx = 0; kx < 9; ++kx) {
        int ix = ox * 2 + kx - 4;
        if (ix >= 0 && ix <= 27) acc += xr[ix] * wr[kx];
      }
    }
    float r = fmaxf(acc, 0.f);
    Ab[o] = r;
    lmax = fmaxf(lmax, r);
  }
  red[tid] = lmax;
  __syncthreads();
  for (int s = 128; s > 0; s >>= 1) {
    if (tid < s) red[tid] = fmaxf(red[tid], red[tid + s]);
    __syncthreads();
  }
  if (tid == 0) gmax[b] = red[0];
}

// ===================== Kernel 2: per-sample kth-largest (radix histogram) ===
__global__ __launch_bounds__(256) void k_thresh(const float* __restrict__ A,
                                                float* __restrict__ thresh) {
  __shared__ unsigned hist[2048];
  __shared__ unsigned selb1, selb2, selb3, selk;
  const int b = blockIdx.x, tid = threadIdx.x;
  const float* base = A + (size_t)b * CHW;

  // pass 1: bits [30:20]
  for (int i = tid; i < 2048; i += 256) hist[i] = 0;
  __syncthreads();
  for (int i = tid; i < CHW; i += 256) {
    __builtin_prefetch(base + i + 4096, 0, 0);
    unsigned u = __float_as_uint(base[i]);
    atomicAdd(&hist[u >> 20], 1u);
  }
  __syncthreads();
  if (tid == 0) {
    unsigned k = KGLOB; int sel = 0;
    for (int bin = 2047; bin >= 0; --bin) {
      unsigned c = hist[bin];
      if (c >= k) { sel = bin; break; }
      k -= c;
    }
    selb1 = (unsigned)sel; selk = k;
  }
  __syncthreads();
  unsigned b1 = selb1;

  // pass 2: bits [19:9]
  for (int i = tid; i < 2048; i += 256) hist[i] = 0;
  __syncthreads();
  for (int i = tid; i < CHW; i += 256) {
    unsigned u = __float_as_uint(base[i]);
    if ((u >> 20) == b1) atomicAdd(&hist[(u >> 9) & 2047u], 1u);
  }
  __syncthreads();
  if (tid == 0) {
    unsigned k = selk; int sel = 0;
    for (int bin = 2047; bin >= 0; --bin) {
      unsigned c = hist[bin];
      if (c >= k) { sel = bin; break; }
      k -= c;
    }
    selb2 = (unsigned)sel; selk = k;
  }
  __syncthreads();
  unsigned b2 = selb2;

  // pass 3: bits [8:0]
  for (int i = tid; i < 2048; i += 256) hist[i] = 0;
  __syncthreads();
  unsigned pref = (b1 << 11) | b2;
  for (int i = tid; i < CHW; i += 256) {
    unsigned u = __float_as_uint(base[i]);
    if ((u >> 9) == pref) atomicAdd(&hist[u & 511u], 1u);
  }
  __syncthreads();
  if (tid == 0) {
    unsigned k = selk; int sel = 0;
    for (int bin = 511; bin >= 0; --bin) {
      unsigned c = hist[bin];
      if (c >= k) { sel = bin; break; }
      k -= c;
    }
    selb3 = (unsigned)sel;
    thresh[b] = __uint_as_float((b1 << 20) | (b2 << 9) | selb3);
  }
}

// ===================== Kernel 3: top-10 scatter + normalize + embed + PE ====
__global__ __launch_bounds__(256) void k_embed(const float* __restrict__ A,
                                               const float* __restrict__ thresh,
                                               const float* __restrict__ gmax,
                                               const float* __restrict__ CE,
                                               float* __restrict__ seq) {
  __shared__ float ce[C1 * DMODEL];
  const int tid = threadIdx.x;
#ifdef USE_TDM
  if (tid < 32) {
    tdm_copy_dwords(CE, lds_offset(ce), (unsigned)(C1 * DMODEL));
    wait_tensor0();
  }
#else
  for (int i = tid; i < C1 * DMODEL; i += 256) ce[i] = CE[i];
#endif
  __syncthreads();

  const int id = blockIdx.x * 256 + tid;     // 784 blocks * 256 = 1024*196
  const int b  = id / SEQLEN;
  const int s  = id - b * SEQLEN;
  const int hh = s / 14;
  const int ww = s - hh * 14;
  const float th  = thresh[b];
  const float gm  = gmax[b];
  const float inv = (gm > 0.f) ? (1.f / gm) : 0.f;
  const float* base = A + (size_t)b * CHW + s;

  float tv[10]; int ti[10];
#pragma unroll
  for (int j = 0; j < 10; ++j) { tv[j] = 0.f; ti[j] = 0; }
  float mn = 0.f; int slot = 0;
  for (int c = 0; c < C1; ++c) {
    float v = base[c * OHW];
    v = (v >= th) ? v : 0.f;
    if (v > mn) {
#pragma unroll
      for (int j = 0; j < 10; ++j)
        if (j == slot) { tv[j] = v; ti[j] = c; }
      mn = tv[0]; slot = 0;
#pragma unroll
      for (int j = 1; j < 10; ++j)
        if (tv[j] < mn) { mn = tv[j]; slot = j; }
    }
  }
  float wv[10];
#pragma unroll
  for (int j = 0; j < 10; ++j) wv[j] = tv[j] * inv;

  // 2D sinusoidal PE: div_j = 10^(-j/2)
  const float divs[8] = {1.f, 0.31622776601683794f, 0.1f, 0.031622776601683794f,
                         0.01f, 0.0031622776601683794f, 0.001f,
                         0.00031622776601683794f};
  float py[16], px[16];
#pragma unroll
  for (int j = 0; j < 8; ++j) {
    float sy, cy, sx, cx;
    __sincosf((float)hh * divs[j], &sy, &cy);
    __sincosf((float)ww * divs[j], &sx, &cx);
    py[2 * j] = sy; py[2 * j + 1] = cy;
    px[2 * j] = sx; px[2 * j + 1] = cx;
  }

  float* dst = seq + (size_t)id * DMODEL;
#pragma unroll
  for (int d = 0; d < DMODEL; ++d) {
    float acc = (d < 16) ? py[d] : px[d - 16];
#pragma unroll
    for (int j = 0; j < 10; ++j) acc += wv[j] * ce[ti[j] * DMODEL + d];
    dst[d] = acc;
  }
}

// ===================== Kernel 4: fused MHA (WMMA + flash softmax) ===========
// LDS map (bytes):
//  [0,13312)       Q  f16 [208][32] row-major
//  [13312,26624)   K  f16 [208][32] row-major
//  [26624,40960)   Vt f16 [32][224] transposed (keys padded/zeroed to 224)
//  [40960,54272)   X  f16 [208][32]  (phase A: seq, phase B/C: ctx)
//  [54272,62464)   Y: phase A ipw f16 [96][32] + biasIn f32[96]; phase B pbuf
//  [62464,64512)   Wout f16 [32][32]
//  [64512,64640)   biasOut f32 [32]
__global__ __launch_bounds__(256) void k_attn(const float* __restrict__ seqg,
                                              const float* __restrict__ ipw,
                                              const float* __restrict__ ipb,
                                              const float* __restrict__ opw,
                                              const float* __restrict__ opb,
                                              float* __restrict__ out) {
  __shared__ __align__(16) unsigned char smem[64640];
  _Float16* Qh    = (_Float16*)(smem);
  _Float16* Kh    = (_Float16*)(smem + 13312);
  _Float16* Vt    = (_Float16*)(smem + 26624);
  _Float16* Xh    = (_Float16*)(smem + 40960);
  _Float16* Yh    = (_Float16*)(smem + 54272);
  float*    bIn   = (float*)   (smem + 54272 + 6144);
  _Float16* Wouth = (_Float16*)(smem + 62464);
  float*    bOut  = (float*)   (smem + 64512);

  const int bidx = blockIdx.x, tid = threadIdx.x;
  // readfirstlane: force the wave index into an SGPR so all task indices are
  // scalar -> fast/tail dispatch compiles to s_cbranch, EXEC stays all-ones.
  const int wave = __builtin_amdgcn_readfirstlane(tid >> 5);
  const int lane = tid & 31;
  const int hi8 = (lane >> 4) << 3;
  const int n15 = lane & 15;
  const bool lo16 = lane < 16;

  // ---- Phase A: zero Q/K/Vt, stage seq/W/biases into LDS ------------------
  unsigned* zp = (unsigned*)smem;
  for (int i = tid; i < 10240; i += 256) zp[i] = 0u;  // 40960 bytes
  {
    const float* sg = seqg + (size_t)bidx * (SEQLEN * DMODEL);
    for (int i = tid; i < 6272; i += 256) Xh[i] = (_Float16)sg[i];
    for (int i = 6272 + tid; i < 6656; i += 256) Xh[i] = (_Float16)0.0f;
  }
  for (int i = tid; i < 3072; i += 256) Yh[i] = (_Float16)ipw[i];   // [96][32]
  if (tid < 96) bIn[tid] = ipb[tid];
  for (int i = tid; i < 1024; i += 256) Wouth[i] = (_Float16)opw[i]; // [32][32]
  if (tid < 32) bOut[tid] = opb[tid];
  __syncthreads();

  // ---- Phase A2: QKV = seq @ W^T + b  (13 x 6 tiles of 16x16, K=32) -------
  // Fast path: qt <= 11 (all rows valid). Tail: qt == 12 (rows 192..207,
  // only rows 192..195 are real -> hi8==0 && r<4).
  for (int t = wave; t < 78; t += 8) {
    int qt = t / 6, nt = t - qt * 6;          // scalar
    v16h a  = frag_a(Xh + qt * 16 * DMODEL, DMODEL, lane);
    v16h bw = frag_b_rowmajor(Yh, nt * 16, lane);
    v8f zc = {};
    v8f d = wmma16(a, bw, zc);
    int col = nt * 16 + n15;
    float bi = bIn[col];
    bool tail = qt == 12;                     // scalar
    if (nt < 4) {                             // scalar: Q or K region
      float scl = (nt < 2) ? QSCALE : 1.f;
      int ridx = (col >> 5) * 6656 + (col & 31) + (qt * 16 + hi8) * 32;
      if (!tail) {
#pragma unroll
        for (int r = 0; r < 8; ++r)
          Qh[ridx + r * 32] = (_Float16)((d[r] + bi) * scl);
      } else {
#pragma unroll
        for (int r = 0; r < 4; ++r)
          if (lo16) Qh[ridx + r * 32] = (_Float16)((d[r] + bi) * scl);
      }
    } else {                                  // scalar: V, transposed store
      int vcol = col - 64;
      if (!tail) {                            // 8 contiguous keys -> one b128
        v8h vv;
#pragma unroll
        for (int r = 0; r < 8; ++r) vv[r] = (_Float16)(d[r] + bi);
        *(v8h*)(Vt + vcol * 224 + qt * 16 + hi8) = vv;
      } else {
#pragma unroll
        for (int r = 0; r < 4; ++r)
          if (lo16) Vt[vcol * 224 + 192 + r] = (_Float16)(d[r] + bi);
      }
    }
  }
  __syncthreads();

  // ---- Phase B: flash attention, tasks = (qt, head). qt==12 is the tail. --
  _Float16* pb = Yh + wave * 512;   // per-wave 16x32 f16 staging tile
  for (int t = wave; t < 52; t += 8) {
    int qt = t >> 2, h = t & 3;               // scalar
    int h8 = h * 8, qt16 = qt * 16;
    v16h aq = frag_a(Qh + qt16 * DMODEL, DMODEL, lane);
    float mr[8], lr[8];
    v8f acc = {};
#pragma unroll
    for (int r = 0; r < 8; ++r) { mr[r] = -1e30f; lr[r] = 0.f; }

    for (int it = 0; it < 6; ++it)            // keys 0..191: no masking needed
      flash_step<false>(Kh, Vt, pb, aq, it * 32, h8, lane, n15, hi8, mr, lr, acc);
    flash_step<true>(Kh, Vt, pb, aq, 192, h8, lane, n15, hi8, mr, lr, acc);

    if (qt < 12) {                            // scalar: all rows valid
#pragma unroll
      for (int r = 0; r < 8; ++r) {
        if (n15 < 8) {
          float o = (lr[r] > 0.f) ? (acc[r] / lr[r]) : 0.f;
          Xh[(qt16 + r + hi8) * DMODEL + h8 + n15] = (_Float16)o;
        }
      }
    } else {                                  // rows 192..195 only
#pragma unroll
      for (int r = 0; r < 4; ++r) {
        if (lo16 && n15 < 8) {
          float o = (lr[r] > 0.f) ? (acc[r] / lr[r]) : 0.f;
          Xh[(192 + r) * DMODEL + h8 + n15] = (_Float16)o;
        }
      }
    }
  }
  __syncthreads();

  // ---- Phase C: out = ctx @ Wout^T + b  (13 x 2 tiles) --------------------
  for (int t = wave; t < 26; t += 8) {
    int qt = t >> 1, nt = t & 1;              // scalar
    v16h a  = frag_a(Xh + qt * 16 * DMODEL, DMODEL, lane);
    v16h bw = frag_b_rowmajor(Wouth, nt * 16, lane);
    v8f zc = {};
    v8f d = wmma16(a, bw, zc);
    int col = nt * 16 + n15;
    float bo = bOut[col];
    float* op = out + ((size_t)bidx * SEQLEN + qt * 16 + hi8) * DMODEL + col;
    if (qt < 12) {                            // scalar: all rows valid
#pragma unroll
      for (int r = 0; r < 8; ++r) op[r * DMODEL] = d[r] + bo;
    } else {                                  // rows 192..195 only
#pragma unroll
      for (int r = 0; r < 4; ++r)
        if (lo16) op[r * DMODEL] = d[r] + bo;
    }
  }
}

// ===================== Host launcher ========================================
extern "C" void kernel_launch(void* const* d_in, const int* in_sizes, int n_in,
                              void* d_out, int out_size, void* d_ws,
                              size_t ws_size, hipStream_t stream) {
  const float* x       = (const float*)d_in[0];
  const float* conv1_w = (const float*)d_in[1];
  const float* ch_emb  = (const float*)d_in[2];
  const float* ipw     = (const float*)d_in[3];
  const float* ipb     = (const float*)d_in[4];
  const float* opw     = (const float*)d_in[5];
  const float* opb     = (const float*)d_in[6];
  float* out = (float*)d_out;

  char* ws = (char*)d_ws;
  const size_t offThresh = (size_t)1024 * CHW * sizeof(float);
  const size_t offGmax   = offThresh + 4096;
  const size_t offSeq    = offGmax + 4096;
  float* A      = (float*)(ws);
  float* thresh = (float*)(ws + offThresh);
  float* gmax   = (float*)(ws + offGmax);
  float* seq    = (float*)(ws + offSeq);

  k_conv  <<<1024, 256, 0, stream>>>(x, conv1_w, A, gmax);
  k_thresh<<<1024, 256, 0, stream>>>(A, thresh);
  k_embed <<<784,  256, 0, stream>>>(A, thresh, gmax, ch_emb, seq);
  k_attn  <<<1024, 256, 0, stream>>>(seq, ipw, ipb, opw, opb, out);
}
